// MultiHeadAttention_56281251446839
// MI455X (gfx1250) — matrix operand order repaired and checked by reading, not compiled, not run on hardware
//
#include <hip/hip_runtime.h>
#include <hip/hip_bf16.h>
#include <math.h>

// ---------------- types ----------------
typedef __attribute__((ext_vector_type(16))) _Float16 v16h;
typedef __attribute__((ext_vector_type(8)))  _Float16 v8h;
typedef __attribute__((ext_vector_type(4)))  _Float16 v4h;
typedef __attribute__((ext_vector_type(8)))  float    v8f;

#define HID   1024
#define HEADS 16
#define HDIM  64
#define BATCH 4
#define SEQ   2048
#define ROWS  (BATCH * SEQ)          // 8192

static __device__ __forceinline__ v16h make_frag(const _Float16* p0,
                                                 const _Float16* p1) {
  v8h a = *(const v8h*)p0;           // global_load_b128 / ds_load_b128
  v8h b = *(const v8h*)p1;
  v16h r;
#pragma unroll
  for (int i = 0; i < 8; ++i) { r[i] = a[i]; r[i + 8] = b[i]; }
  return r;
}

static __device__ __forceinline__ v8f wmma_f16(v16h a, v16h b, v8f c) {
  return __builtin_amdgcn_wmma_f32_16x16x32_f16(false, a, false, b,
                                                (short)0, c, false, false);
}

// Same-wave LDS ordering: DS ops are in-order per wave (ISA 7.3); this fence
// stops compiler reordering and drains DScnt without a workgroup barrier.
static __device__ __forceinline__ void lds_fence() {
  asm volatile("s_wait_dscnt 0x0" ::: "memory");
}

// ---------------- 1) f32 -> f16 convert of hidden_states ----------------
__global__ void cvt_x_kernel(const float4* __restrict__ in,
                             _Float16* __restrict__ out, int n4) {
  int i = blockIdx.x * blockDim.x + threadIdx.x;
  if (i >= n4) return;
  float4 v = in[i];
  v4h h;
  h[0] = (_Float16)v.x; h[1] = (_Float16)v.y;
  h[2] = (_Float16)v.z; h[3] = (_Float16)v.w;
  *(v4h*)(out + 4 * (size_t)i) = h;
}

// ---------------- 2) transpose + convert weights: Wt[n][k] = W[k][n] -------
__global__ void transpose_w_kernel(const float* __restrict__ Wq,
                                   const float* __restrict__ Wk,
                                   const float* __restrict__ Wv,
                                   _Float16* __restrict__ Wt) {
  __shared__ float tile[32][33];
  const int w = blockIdx.z;
  const float* W = (w == 0) ? Wq : ((w == 1) ? Wk : Wv);
  const int kbase = blockIdx.y * 32;   // row of W (input dim)
  const int nbase = blockIdx.x * 32;   // col of W (output dim)
#pragma unroll
  for (int i = threadIdx.y; i < 32; i += 8)
    tile[i][threadIdx.x] = W[(size_t)(kbase + i) * HID + nbase + threadIdx.x];
  __syncthreads();
  _Float16* O = Wt + (size_t)w * HID * HID;
#pragma unroll
  for (int i = threadIdx.y; i < 32; i += 8)
    O[(size_t)(nbase + i) * HID + kbase + threadIdx.x] =
        (_Float16)tile[threadIdx.x][i];
}

// ---------------- 3) fused QKV GEMM (WMMA, 32x64 tile) ----------------
// One wave: 32x64 tile of one of {Q,K,V}: 8 WMMAs per k-step, A reused 4x and B
// reused twice.  No manual double-buffer: backend pipelines the b128 clauses
// itself (round-1 codegen) without spilling the 64 accumulator VGPRs.
__global__ void qkv_gemm_kernel(const _Float16* __restrict__ Xh,
                                const _Float16* __restrict__ Wt,
                                const float* __restrict__ bq,
                                const float* __restrict__ bk,
                                const float* __restrict__ bv,
                                _Float16* __restrict__ Qb,
                                _Float16* __restrict__ Kb,
                                _Float16* __restrict__ Vt) {
  const int wave = blockIdx.x * (blockDim.x >> 5) + (threadIdx.x >> 5);
  const int lane = threadIdx.x & 31;
  const int mtile = wave & 255;              // 256 M tiles of 32 rows
  const int rest  = wave >> 8;
  const int nt64  = rest & 15;               // 16 N64 tiles
  const int which = rest >> 4;               // 0=Q 1=K 2=V
  if (which >= 3) return;

  const _Float16* W = Wt + (size_t)which * HID * HID;
  const float* bias = (which == 0) ? bq : ((which == 1) ? bk : bv);

  const int m0 = mtile * 32;
  const int n0 = nt64 * 64;
  const int ncol = lane & 15;
  const int aoff = (lane & 16) ? 16 : 0;     // A-frag k sub-offset
  const int boff = (lane & 16) ? 16 : 0;     // B-frag k sub-offset

  const _Float16* arow0 = Xh + (size_t)(m0 + ncol) * HID;
  const _Float16* arow1 = arow0 + (size_t)16 * HID;
  const _Float16* brow[4];
#pragma unroll
  for (int t = 0; t < 4; ++t)
    brow[t] = W + (size_t)(n0 + t * 16 + ncol) * HID;

  v8f c[2][4] = {{{}, {}, {}, {}}, {{}, {}, {}, {}}};

#pragma unroll 2
  for (int k0 = 0; k0 < HID; k0 += 32) {
    if (k0 + 64 < HID)
      __builtin_prefetch(arow0 + k0 + 64, 0, 1);   // global_prefetch_b8
    const v16h a0 = make_frag(arow0 + k0 + aoff, arow0 + k0 + aoff + 8);
    const v16h a1 = make_frag(arow1 + k0 + aoff, arow1 + k0 + aoff + 8);
#pragma unroll
    for (int t = 0; t < 4; ++t) {
      const v16h bt = make_frag(brow[t] + k0 + boff, brow[t] + k0 + boff + 8);
      c[0][t] = wmma_f16(a0, bt, c[0][t]);
      c[1][t] = wmma_f16(a1, bt, c[1][t]);
    }
  }

  const int rowhi = (lane & 16) ? 8 : 0;
#pragma unroll
  for (int mi = 0; mi < 2; ++mi) {
#pragma unroll
    for (int t = 0; t < 4; ++t) {
      const int n = n0 + t * 16 + ncol;
      const float bi = bias[n];
#pragma unroll
      for (int r = 0; r < 8; ++r) {
        const int m = m0 + mi * 16 + rowhi + r;
        float v = c[mi][t][r] + bi;
        if (which == 0) {
          Qb[(size_t)m * HID + n] = (_Float16)(v * 0.125f); // fold 1/sqrt(64)
        } else if (which == 1) {
          Kb[(size_t)m * HID + n] = (_Float16)v;
        } else {
          const int bb = m >> 11, s = m & 2047;             // V transposed
          Vt[((size_t)bb * HID + n) * SEQ + s] = (_Float16)v;
        }
      }
    }
  }
}

// ---------------- 4) flash attention (WMMA + online softmax) ----------------
#define WPB 8   // waves per block (each wave fully independent)
__global__ void attn_kernel(const _Float16* __restrict__ Qb,
                            const _Float16* __restrict__ Kb,
                            const _Float16* __restrict__ Vt,
                            float* __restrict__ out) {
  __shared__ __align__(16) _Float16 lds[WPB][16 * 32];
  const int wavein = threadIdx.x >> 5;
  const int wave   = blockIdx.x * WPB + wavein;
  const int lane   = threadIdx.x & 31;

  const int qblk = wave & 127;               // 128 query blocks of 16
  const int bh   = wave >> 7;                // 0..63
  const int b = bh >> 4, h = bh & 15;
  const int q0 = qblk * 16;

  const int ncol = lane & 15;
  const int aoff = (lane & 16) ? 16 : 0;
  const int bo   = (lane & 16) ? 16 : 0;

  // Q A-fragments (d = 0..31 and 32..63), already scaled by 1/8
  const _Float16* Qrow =
      Qb + ((size_t)b * SEQ + q0 + ncol) * HID + h * HDIM;
  const v16h aq0 = make_frag(Qrow + aoff,      Qrow + aoff + 8);
  const v16h aq1 = make_frag(Qrow + 32 + aoff, Qrow + 32 + aoff + 8);

  float mrow[8], lrow[8];
#pragma unroll
  for (int r = 0; r < 8; ++r) { mrow[r] = -__builtin_inff(); lrow[r] = 0.f; }
  v8f acc[4] = {{}, {}, {}, {}};

  _Float16* L = lds[wavein];
  const int prow = (lane & 16) ? 8 : 0;      // C-layout row base
  const int pm   = lane & 15;                // A-layout row

  for (int j0 = 0; j0 < SEQ; j0 += 32) {
    // ---- scores: two 16x16 column tiles, K-dim = head_dim (2 steps) ----
    const _Float16* Krow0 =
        Kb + ((size_t)b * SEQ + j0 + ncol) * HID + h * HDIM;
    const _Float16* Krow1 = Krow0 + (size_t)16 * HID;
    v8f c0 = {}, c1 = {};
    c0 = wmma_f16(aq0, make_frag(Krow0 + bo,      Krow0 + bo + 8),      c0);
    c0 = wmma_f16(aq1, make_frag(Krow0 + 32 + bo, Krow0 + 32 + bo + 8), c0);
    c1 = wmma_f16(aq0, make_frag(Krow1 + bo,      Krow1 + bo + 8),      c1);
    c1 = wmma_f16(aq1, make_frag(Krow1 + 32 + bo, Krow1 + 32 + bo + 8), c1);

    // ---- V B-fragments issued early: latency hides under softmax VALU ----
    v16h vb[4];
#pragma unroll
    for (int t = 0; t < 4; ++t) {
      const _Float16* vp =
          Vt + ((size_t)b * HID + h * HDIM + t * 16 + ncol) * SEQ + j0 + bo;
      vb[t] = make_frag(vp, vp + 8);
    }

    // ---- online softmax (rows live across 16-lane halves) ----
    float scale[8];
#pragma unroll
    for (int r = 0; r < 8; ++r) {
      float v = fmaxf(c0[r], c1[r]);
      v = fmaxf(v, __shfl_xor(v, 1, 32));
      v = fmaxf(v, __shfl_xor(v, 2, 32));
      v = fmaxf(v, __shfl_xor(v, 4, 32));
      v = fmaxf(v, __shfl_xor(v, 8, 32));
      const float mn = fmaxf(mrow[r], v);
      const float sc = __expf(mrow[r] - mn);
      const float p0 = __expf(c0[r] - mn);
      const float p1 = __expf(c1[r] - mn);
      c0[r] = p0; c1[r] = p1;
      float s = p0 + p1;
      s += __shfl_xor(s, 1, 32);
      s += __shfl_xor(s, 2, 32);
      s += __shfl_xor(s, 4, 32);
      s += __shfl_xor(s, 8, 32);
      lrow[r] = lrow[r] * sc + s;
      mrow[r] = mn;
      scale[r] = sc;
    }
#pragma unroll
    for (int t = 0; t < 4; ++t)
#pragma unroll
      for (int r = 0; r < 8; ++r) acc[t][r] *= scale[r];

    // ---- reshape P: C-layout -> A-layout through this wave's LDS strip ----
#pragma unroll
    for (int r = 0; r < 8; ++r) {
      L[(prow + r) * 32 + ncol]      = (_Float16)c0[r];
      L[(prow + r) * 32 + 16 + ncol] = (_Float16)c1[r];
    }
    lds_fence();   // same-wave DS in-order; just block compiler reordering
    const v16h pa = make_frag(L + pm * 32 + aoff, L + pm * 32 + aoff + 8);
    lds_fence();   // loads retired before next iteration's overwrite

    // ---- PV accumulate ----
#pragma unroll
    for (int t = 0; t < 4; ++t)
      acc[t] = wmma_f16(pa, vb[t], acc[t]);
  }

  // ---- normalize + store context [B,S,HID] f32 ----
#pragma unroll
  for (int t = 0; t < 4; ++t) {
#pragma unroll
    for (int r = 0; r < 8; ++r) {
      const int m = q0 + prow + r;
      out[((size_t)b * SEQ + m) * HID + h * HDIM + t * 16 + ncol] =
          acc[t][r] / lrow[r];
    }
  }
}

// ---------------- launcher ----------------
extern "C" void kernel_launch(void* const* d_in, const int* in_sizes, int n_in,
                              void* d_out, int out_size, void* d_ws,
                              size_t ws_size, hipStream_t stream) {
  const float* X  = (const float*)d_in[0];
  const float* Wq = (const float*)d_in[1];
  const float* bq = (const float*)d_in[2];
  const float* Wk = (const float*)d_in[3];
  const float* bk = (const float*)d_in[4];
  const float* Wv = (const float*)d_in[5];
  const float* bv = (const float*)d_in[6];
  float* out = (float*)d_out;

  // workspace carve (all f16)
  _Float16* Xh = (_Float16*)d_ws;                       // 8192*1024
  _Float16* Wt = Xh + (size_t)ROWS * HID;               // 3*1024*1024
  _Float16* Qb = Wt + (size_t)3 * HID * HID;            // 8192*1024
  _Float16* Kb = Qb + (size_t)ROWS * HID;               // 8192*1024
  _Float16* Vt = Kb + (size_t)ROWS * HID;               // 4*1024*2048

  // 1) convert hidden_states
  {
    const int n4 = ROWS * HID / 4;
    cvt_x_kernel<<<(n4 + 255) / 256, 256, 0, stream>>>((const float4*)X, Xh, n4);
  }
  // 2) transpose + convert weights
  transpose_w_kernel<<<dim3(HID / 32, HID / 32, 3), dim3(32, 8), 0, stream>>>(
      Wq, Wk, Wv, Wt);
  // 3) fused QKV GEMM: 256 * 16 * 3 = 12288 waves, 8 waves/block
  qkv_gemm_kernel<<<12288 / 8, 256, 0, stream>>>(Xh, Wt, bq, bk, bv, Qb, Kb, Vt);
  // 4) attention: 64 bh * 128 qblocks = 8192 waves, 8 waves/block
  attn_kernel<<<8192 / WPB, WPB * 32, 0, stream>>>(Qb, Kb, Vt, out);
}
